// FeatureMatching_48034914238597
// MI455X (gfx1250) — matrix operand order repaired and checked by reading, not compiled, not run on hardware
//
#include <hip/hip_runtime.h>
#include <stdint.h>

typedef __attribute__((ext_vector_type(16))) _Float16 v16h;
typedef __attribute__((ext_vector_type(8)))  float    v8f;

#define H0 192
#define W0 192
#define HC 96
#define WC 96
#define NPOS (HC*WC)            /* 9216 */
#define KF 144
#define KP 160                  /* padded K = 5 * 32 */
#define LDS_STRIDE 168          /* halves per LDS row: 336B, 16B aligned */
#define CHUNK2 64               /* ref rows per LDS buffer (double buffered) */
#define NCHUNK2 (NPOS/CHUNK2)   /* 144 */
#define NWAVE 8
#define COLS_PER_BLOCK (NWAVE*16) /* 128 */
#define NBLK (NPOS/COLS_PER_BLOCK) /* 72 */

__constant__ float c_mean[3] = {0.485f, 0.456f, 0.406f};
__constant__ float c_std[3]  = {0.229f, 0.224f, 0.225f};

// ---------------- preprocessing ----------------
__global__ void k_submean_lr(const float* __restrict__ x, float* __restrict__ y) {
    int i = blockIdx.x * blockDim.x + threadIdx.x;
    if (i >= 3 * H0 * W0) return;
    int c = i / (H0 * W0);
    y[i] = (x[i] - c_mean[c]) / c_std[c];
}

__global__ void k_avg_submean_ref(const float* __restrict__ x, float* __restrict__ y) {
    int i = blockIdx.x * blockDim.x + threadIdx.x;
    if (i >= 3 * H0 * W0) return;
    int c = i / (H0 * W0), rem = i % (H0 * W0);
    int yy = rem / W0, xx = rem % W0;
    const float* p = x + ((size_t)c * 384 + yy * 2) * 384 + xx * 2;
    float v = 0.25f * (p[0] + p[1] + p[384] + p[385]);
    y[i] = (v - c_mean[c]) / c_std[c];
}

// ---------------- conv1 only (Cin=3): direct f32, 8 co per block ----------------
__global__ __launch_bounds__(256) void k_conv3x3(
    const float* __restrict__ in, const float* __restrict__ wgt,
    const float* __restrict__ bias, float* __restrict__ out,
    int Cin, int H, int W) {
    __shared__ float tile[18][19];
    int tx = threadIdx.x, ty = threadIdx.y;
    int x0 = blockIdx.x * 16, y0 = blockIdx.y * 16;
    int co0 = blockIdx.z * 8;
    float acc[8] = {0.f, 0.f, 0.f, 0.f, 0.f, 0.f, 0.f, 0.f};
    for (int ci = 0; ci < Cin; ++ci) {
        for (int idx = ty * 16 + tx; idx < 18 * 18; idx += 256) {
            int ly = idx / 18, lx = idx % 18;
            int gy = y0 + ly - 1, gx = x0 + lx - 1;
            float v = 0.f;
            if ((unsigned)gy < (unsigned)H && (unsigned)gx < (unsigned)W)
                v = in[((size_t)ci * H + gy) * W + gx];
            tile[ly][lx] = v;
        }
        __syncthreads();
        const float* wp = wgt + ((size_t)co0 * Cin + ci) * 9;
        #pragma unroll
        for (int ky = 0; ky < 3; ky++)
            #pragma unroll
            for (int kx = 0; kx < 3; kx++) {
                float p = tile[ty + ky][tx + kx];
                #pragma unroll
                for (int o = 0; o < 8; o++)
                    acc[o] = fmaf(p, wp[(size_t)o * Cin * 9 + ky * 3 + kx], acc[o]);
            }
        __syncthreads();
    }
    int y = y0 + ty, x = x0 + tx;
    #pragma unroll
    for (int o = 0; o < 8; o++) {
        float v = acc[o] + bias[co0 + o];
        out[((size_t)(co0 + o) * H + y) * W + x] = fmaxf(v, 0.f);
    }
}

// ---- weight reorder for WMMA conv: w16[co][s*Cin+ci] = W[co][ci][s] (f16) ----
__global__ void k_wprep(const float* __restrict__ w, _Float16* __restrict__ w16,
                        int Cout, int Cin) {
    int i = blockIdx.x * blockDim.x + threadIdx.x;
    if (i >= Cout * Cin * 9) return;
    int co = i / (Cin * 9), r = i % (Cin * 9);
    int s = r / Cin, ci = r % Cin;
    w16[i] = (_Float16)w[((size_t)co * Cin + ci) * 9 + s];
}

// ---- NCHW f32 -> zero-padded NHWC f16: out[(H+2)][(W+2)][C] ----
__global__ void k_nhwc_pad(const float* __restrict__ in, _Float16* __restrict__ out,
                           int C, int H, int W) {
    int i = blockIdx.x * blockDim.x + threadIdx.x;
    int Wp = W + 2;
    if (i >= (H + 2) * Wp * C) return;
    int ci = i % C, rem = i / C;
    int xx = rem % Wp, yy = rem / Wp;
    int sy = yy - 1, sx = xx - 1;
    float v = 0.f;
    if ((unsigned)sy < (unsigned)H && (unsigned)sx < (unsigned)W)
        v = in[((size_t)ci * H + sy) * W + sx];
    out[i] = (_Float16)v;
}

// ---- implicit-GEMM 3x3 conv on WMMA; CIN/H/W compile-time, full K unroll ----
// Block: 8 waves = 4 (co tiles of 16) x 2 (pos tiles of 16); tile 64co x 32pos.
template <int CIN, int H, int W>
__global__ __launch_bounds__(256) void k_conv_wmma(
    const _Float16* __restrict__ actp,  // (H+2)x(W+2)xCIN f16 (zero padded)
    const _Float16* __restrict__ w16,   // Cout x (9*CIN) f16, K = s*CIN+ci
    const float* __restrict__ bias,
    float* __restrict__ out) {          // Cout x H x W f32
    const int tid = threadIdx.x;
    const int wave = tid >> 5;
    const int lane = tid & 31;
    const int n = lane & 15, hi = lane >> 4;
    const int mrow = wave >> 1;          // 0..3
    const int ncol = wave & 1;           // 0..1
    const int co0 = blockIdx.y * 64 + mrow * 16;
    const int pos = blockIdx.x * 32 + ncol * 16 + n;   // 32 | W -> single row
    const int y = pos / W, x = pos % W;
    constexpr int Wp = W + 2;
    constexpr int Ktot = 9 * CIN;
    constexpr int ksteps = Ktot / 32;
    union HU { v16h h; uint4 q[2]; };
    v8f c = {};
    const _Float16* arow = w16 + (size_t)(co0 + n) * Ktot + hi * 8;  // A: M = lane&15
    const _Float16* brow = actp + ((size_t)y * Wp + x) * CIN + hi * 8;
    #pragma unroll
    for (int ks = 0; ks < ksteps; ++ks) {
        constexpr int per_s = CIN / 32;       // k-step chunks per tap
        const int s = ks / per_s;             // compile-time per unrolled iter
        const int ci0 = (ks % per_s) * 32;
        const int dy = s / 3, dx = s % 3;
        HU A, B;
        const _Float16* ap = arow + ks * 32;
        A.q[0] = *(const uint4*)(ap);
        A.q[1] = *(const uint4*)(ap + 16);
        const _Float16* bp = brow + ((size_t)dy * Wp + dx) * CIN + ci0;
        B.q[0] = *(const uint4*)(bp);
        B.q[1] = *(const uint4*)(bp + 16);
        c = __builtin_amdgcn_wmma_f32_16x16x32_f16(
                false, A.h, false, B.h, (short)0, c, false, false);
    }
    #pragma unroll
    for (int i = 0; i < 8; i++) {
        int co = co0 + i + 8 * hi;           // C layout: VGPR i -> M = i + 8*hi
        float v = c[i] + bias[co];
        out[(size_t)co * H * W + pos] = fmaxf(v, 0.f);
    }
}

__global__ void k_maxpool(const float* __restrict__ in, float* __restrict__ out,
                          int C, int H, int W) {
    int ho = H / 2, wo = W / 2;
    int i = blockIdx.x * blockDim.x + threadIdx.x;
    if (i >= C * ho * wo) return;
    int c = i / (ho * wo), rem = i % (ho * wo);
    int y = rem / wo, x = rem % wo;
    const float* p = in + ((size_t)c * H + 2 * y) * W + 2 * x;
    out[i] = fmaxf(fmaxf(p[0], p[1]), fmaxf(p[W], p[W + 1]));
}

__global__ void k_conv1x1_leaky(const float* __restrict__ in, const float* __restrict__ w,
                                const float* __restrict__ b, float* __restrict__ out) {
    int i = blockIdx.x * blockDim.x + threadIdx.x;
    if (i >= 16 * NPOS) return;
    int co = i / NPOS, p = i % NPOS;
    float acc = b[co];
    #pragma unroll 8
    for (int ci = 0; ci < 128; ci++)
        acc = fmaf(in[(size_t)ci * NPOS + p], w[co * 128 + ci], acc);
    out[i] = acc > 0.f ? acc : 0.2f * acc;
}

// -------- 3x3 patch extract + l2norm + f16 pack: out[pos][KP] --------
__global__ void k_patchnorm(const float* __restrict__ f, _Float16* __restrict__ outp) {
    int p = blockIdx.x * blockDim.x + threadIdx.x;
    if (p >= NPOS) return;
    int y = p / WC, x = p % WC;
    float ss = 0.f;
    for (int c = 0; c < 16; c++)
        #pragma unroll
        for (int s = 0; s < 9; s++) {
            int gy = y + s / 3 - 1, gx = x + s % 3 - 1;
            float v = ((unsigned)gy < HC && (unsigned)gx < WC)
                          ? f[((size_t)c * HC + gy) * WC + gx] : 0.f;
            ss += v * v;
        }
    float inv = 1.f / fmaxf(sqrtf(ss), 1e-12f);
    _Float16* o = outp + (size_t)p * KP;
    for (int c = 0; c < 16; c++)
        #pragma unroll
        for (int s = 0; s < 9; s++) {
            int gy = y + s / 3 - 1, gx = x + s % 3 - 1;
            float v = ((unsigned)gy < HC && (unsigned)gx < WC)
                          ? f[((size_t)c * HC + gy) * WC + gx] : 0.f;
            o[c * 9 + s] = (_Float16)(v * inv);
        }
    #pragma unroll
    for (int k = KF; k < KP; k++) o[k] = (_Float16)0.f;
}

// -------- fused correlation GEMM (WMMA f16) + max/argmax over ref rows --------
// Double-buffered async LDS chunks + software-pipelined A-fragment loads.
__global__ __launch_bounds__(256) void k_corr_argmax(
    const _Float16* __restrict__ refp, const _Float16* __restrict__ lrp,
    float* __restrict__ rel96, float* __restrict__ idx_out) {
    __shared__ _Float16 sA[2][CHUNK2 * LDS_STRIDE];
    const int tid = threadIdx.x;
    const int wave = tid >> 5;
    const int lane = tid & 31;
    const int n = lane & 15;
    const int hi = lane >> 4;
    const int col = blockIdx.x * COLS_PER_BLOCK + wave * 16 + n;

    union HU { v16h h; uint4 q[2]; };

    // B (lr columns) is loop-invariant: 5 k-steps in registers.
    HU B[5];
    #pragma unroll
    for (int s = 0; s < 5; s++) {
        const uint4* p = (const uint4*)(lrp + (size_t)col * KP + s * 32 + hi * 8);
        B[s].q[0] = p[0];
        B[s].q[1] = p[2];
    }

    // async-copy plan: 64 rows x 20 x16B segments = 1280 segs = 5 per thread
    unsigned loff[5], goff[5];
    #pragma unroll
    for (int g = 0; g < 5; ++g) {
        int seg = tid + g * 256;
        int r = seg / 20, s16 = seg % 20;
        loff[g] = (unsigned)(r * (LDS_STRIDE * 2) + s16 * 16);
        goff[g] = (unsigned)(r * (KP * 2) + s16 * 16);
    }
    unsigned lb[2] = { (unsigned)(uintptr_t)&sA[0][0], (unsigned)(uintptr_t)&sA[1][0] };

    auto issue = [&](int ch) {
        const char* gp = (const char*)refp + (size_t)ch * (CHUNK2 * KP * 2);
        unsigned base = lb[ch & 1];
        #pragma unroll
        for (int g = 0; g < 5; ++g) {
            asm volatile("global_load_async_to_lds_b128 %0, %1, off"
                         :: "v"(base + loff[g]), "v"((const void*)(gp + goff[g]))
                         : "memory");
        }
    };

    float best = -1e30f;
    int bidx = 0;

    issue(0);
    for (int ch = 0; ch < NCHUNK2; ++ch) {
        if (ch + 1 < NCHUNK2) {
            issue(ch + 1);
            asm volatile("s_wait_asynccnt 0x5" ::: "memory");  // prev batch done
        } else {
            asm volatile("s_wait_asynccnt 0x0" ::: "memory");
        }
        __syncthreads();

        const _Float16* sbuf = &sA[ch & 1][0];
        HU A[2][5];
        auto ldA = [&](int it, HU* Ad) {
            const _Float16* rowp = sbuf + (it * 16 + n) * LDS_STRIDE + hi * 8;
            #pragma unroll
            for (int s = 0; s < 5; s++) {
                Ad[s].q[0] = *(const uint4*)(rowp + s * 32);
                Ad[s].q[1] = *(const uint4*)(rowp + s * 32 + 16);
            }
        };
        ldA(0, A[0]);
        const int rowbase = ch * CHUNK2;
        #pragma unroll
        for (int it = 0; it < CHUNK2 / 16; ++it) {
            if (it + 1 < CHUNK2 / 16) ldA(it + 1, A[(it + 1) & 1]);
            v8f c = {};
            #pragma unroll
            for (int s = 0; s < 5; s++)
                c = __builtin_amdgcn_wmma_f32_16x16x32_f16(
                        false, A[it & 1][s].h, false, B[s].h, (short)0, c, false, false);
            int rb = rowbase + it * 16 + hi * 8;
            #pragma unroll
            for (int i = 0; i < 8; i++) {
                float v = c[i];
                if (v > best) { best = v; bidx = rb + i; }
            }
        }
        __syncthreads();
    }

    float obest = __shfl_down(best, 16, 32);
    int   oidx  = __shfl_down(bidx, 16, 32);
    if (obest > best || (obest == best && oidx < bidx)) { best = obest; bidx = oidx; }
    if (hi == 0) {
        rel96[col] = best;
        idx_out[col] = (float)bidx;
    }
}

// -------- bicubic (Keys a=-0.5) 96 -> 192 upsample + clamp --------
__device__ inline float cub05(float d) {
    d = fabsf(d);
    if (d <= 1.f) return ((1.5f * d - 2.5f) * d) * d + 1.f;
    if (d < 2.f)  return ((-0.5f * d + 2.5f) * d - 4.f) * d + 2.f;
    return 0.f;
}

__global__ void k_bicubic(const float* __restrict__ s, float* __restrict__ o) {
    int i = blockIdx.x * blockDim.x + threadIdx.x;
    if (i >= H0 * W0) return;
    int oy = i / W0, ox = i % W0;
    float sy = 0.5f * oy - 0.25f, sx = 0.5f * ox - 0.25f;
    int iy = (int)floorf(sy), ix = (int)floorf(sx);
    float acc = 0.f;
    #pragma unroll
    for (int ky = 0; ky < 4; ky++) {
        int yy = min(max(iy - 1 + ky, 0), HC - 1);
        float wy = cub05(sy - (float)(iy - 1 + ky));
        float row = 0.f;
        #pragma unroll
        for (int kx = 0; kx < 4; kx++) {
            int xx = min(max(ix - 1 + kx, 0), WC - 1);
            row = fmaf(cub05(sx - (float)(ix - 1 + kx)), s[yy * WC + xx], row);
        }
        acc = fmaf(wy, row, acc);
    }
    o[i] = fminf(fmaxf(acc, 0.f), 1.f);
}

// ---------------- host side ----------------
extern "C" void kernel_launch(void* const* d_in, const int* in_sizes, int n_in,
                              void* d_out, int out_size, void* d_ws, size_t ws_size,
                              hipStream_t stream) {
    const float* lr  = (const float*)d_in[0];
    const float* ref = (const float*)d_in[1];
    const float* W1 = (const float*)d_in[2];
    const float* b1 = (const float*)d_in[3];
    const float* W2 = (const float*)d_in[4];
    const float* b2 = (const float*)d_in[5];
    const float* W3 = (const float*)d_in[6];
    const float* b3 = (const float*)d_in[7];
    const float* Wm = (const float*)d_in[8];
    const float* bm = (const float*)d_in[9];
    float* out = (float*)d_out;

    size_t off = 0;
    auto carve = [&](size_t bytes) -> void* {
        void* p = (char*)d_ws + off;
        off += (bytes + 255) & ~(size_t)255;
        return p;
    };
    float* p_in  = (float*)carve(3 * H0 * W0 * 4);
    float* t1    = (float*)carve((size_t)64 * H0 * W0 * 4);
    float* t2    = (float*)carve((size_t)64 * H0 * W0 * 4);
    float* t3    = (float*)carve((size_t)64 * HC * WC * 4);
    float* t4    = (float*)carve((size_t)128 * HC * WC * 4);
    float* f_lr  = (float*)carve((size_t)16 * NPOS * 4);
    float* f_ref = (float*)carve((size_t)16 * NPOS * 4);
    _Float16* w16_2 = (_Float16*)carve((size_t)64 * 576 * 2);
    _Float16* w16_3 = (_Float16*)carve((size_t)128 * 576 * 2);
    _Float16* actp2 = (_Float16*)carve((size_t)194 * 194 * 64 * 2);
    _Float16* actp3 = (_Float16*)carve((size_t)98 * 98 * 64 * 2);
    _Float16* lrp   = (_Float16*)carve((size_t)NPOS * KP * 2);
    _Float16* refp  = (_Float16*)carve((size_t)NPOS * KP * 2);
    float* rel96 = (float*)carve(NPOS * 4);
    (void)ws_size; (void)n_in; (void)in_sizes; (void)out_size;

    dim3 blk2(16, 16);

    // weight reorder (once)
    k_wprep<<<(64 * 576 + 255) / 256, 256, 0, stream>>>(W2, w16_2, 64, 64);
    k_wprep<<<(128 * 576 + 255) / 256, 256, 0, stream>>>(W3, w16_3, 128, 64);

    const float* srcs[2] = {lr, ref};
    float* feats[2] = {f_lr, f_ref};
    for (int pass = 0; pass < 2; ++pass) {
        if (pass == 0)
            k_submean_lr<<<(3 * H0 * W0 + 255) / 256, 256, 0, stream>>>(srcs[pass], p_in);
        else
            k_avg_submean_ref<<<(3 * H0 * W0 + 255) / 256, 256, 0, stream>>>(srcs[pass], p_in);
        // conv1 (Cin=3) direct
        k_conv3x3<<<dim3(12, 12, 8), blk2, 0, stream>>>(p_in, W1, b1, t1, 3, H0, W0);
        // conv2 via WMMA implicit GEMM
        k_nhwc_pad<<<(194 * 194 * 64 + 255) / 256, 256, 0, stream>>>(t1, actp2, 64, H0, W0);
        k_conv_wmma<64, H0, W0><<<dim3(1152, 1), 256, 0, stream>>>(actp2, w16_2, b2, t2);
        // maxpool
        k_maxpool<<<(64 * NPOS + 255) / 256, 256, 0, stream>>>(t2, t3, 64, H0, W0);
        // conv3 via WMMA implicit GEMM
        k_nhwc_pad<<<(98 * 98 * 64 + 255) / 256, 256, 0, stream>>>(t3, actp3, 64, HC, WC);
        k_conv_wmma<64, HC, WC><<<dim3(288, 2), 256, 0, stream>>>(actp3, w16_3, b3, t4);
        // 1x1 + leaky
        k_conv1x1_leaky<<<(16 * NPOS + 255) / 256, 256, 0, stream>>>(t4, Wm, bm, feats[pass]);
    }

    // patches + l2 norm + f16 pack
    k_patchnorm<<<(NPOS + 255) / 256, 256, 0, stream>>>(f_lr, lrp);
    k_patchnorm<<<(NPOS + 255) / 256, 256, 0, stream>>>(f_ref, refp);

    // fused WMMA correlation + argmax; indices (as float) -> out[36864..]
    k_corr_argmax<<<NBLK, 256, 0, stream>>>(refp, lrp, rel96, out + H0 * W0);

    // bicubic upsample of relevance map -> out[0..36863]
    k_bicubic<<<(H0 * W0 + 255) / 256, 256, 0, stream>>>(rel96, out);
}
